// NodeAttentionHardmax_32959579030129
// MI455X (gfx1250) — matrix operand order repaired and checked by reading, not compiled, not run on hardware
//
#include <hip/hip_runtime.h>
#include <hip/hip_bf16.h>

#define EPS 1e-5f
#define NEG_SLOPE 0.2f

static constexpr int Nn   = 32;
static constexpr int C    = 1024;
static constexpr int T    = 32;
static constexpr int Hh   = 7;
static constexpr int Ww   = 7;
static constexpr int SPN  = T * Hh * Ww;   // 1568 spatial positions per batch
static constexpr int STOT = Nn * SPN;      // 50176 total rows of the big GEMM
static constexpr int M    = 128;           // nodes

typedef __attribute__((ext_vector_type(16))) __bf16 v16bf;
typedef __attribute__((ext_vector_type(8)))  float  v8f;

// ---------------------------------------------------------------------------
// k1: per-channel (over n,t,h,w) stats of input, folded into affine a1,b1
// phi = x*a1_c + b1_c.  float4 loads -> global_load_b128 on the 205MB stream.
// ---------------------------------------------------------------------------
__global__ void k1_input_stats(const float* __restrict__ x,
                               const float* __restrict__ g1,
                               const float* __restrict__ be1,
                               float* __restrict__ a1, float* __restrict__ b1) {
    __shared__ float ssum[256], ssq[256];
    int c = blockIdx.x, tid = threadIdx.x;
    float s = 0.f, q = 0.f;
    const float* base = x + (size_t)c * SPN;
    for (int n = 0; n < Nn; ++n) {
        const float4* p = (const float4*)(base + (size_t)n * C * SPN);
        for (int i = tid; i < SPN / 4; i += 256) {
            float4 v = p[i];
            s += v.x + v.y + v.z + v.w;
            q += v.x * v.x + v.y * v.y + v.z * v.z + v.w * v.w;
        }
    }
    ssum[tid] = s; ssq[tid] = q; __syncthreads();
    for (int off = 128; off > 0; off >>= 1) {
        if (tid < off) { ssum[tid] += ssum[tid + off]; ssq[tid] += ssq[tid + off]; }
        __syncthreads();
    }
    if (tid == 0) {
        float mean = ssum[0] / (float)STOT;
        float var  = ssq[0]  / (float)STOT - mean * mean;
        float a = rsqrtf(var + EPS) * g1[c];
        a1[c] = a;
        b1[c] = be1[c] - mean * a;
    }
}

// ---------------------------------------------------------------------------
// k2: per-channel (over m) stats of nodes; emits theta0 = bn2(nodes),
// gmat = bn5(nodes). (Same stats, different affine.)
// ---------------------------------------------------------------------------
__global__ void k2_nodes(const float* __restrict__ nodes,
                         const float* __restrict__ g2, const float* __restrict__ be2,
                         const float* __restrict__ g5, const float* __restrict__ be5,
                         float* __restrict__ theta0, float* __restrict__ gmat) {
    int c = blockIdx.x * blockDim.x + threadIdx.x;
    if (c >= C) return;
    float s = 0.f, q = 0.f;
    for (int m = 0; m < M; ++m) { float v = nodes[m * C + c]; s += v; q += v * v; }
    float mean = s / (float)M;
    float var  = q / (float)M - mean * mean;
    float rs   = rsqrtf(var + EPS);
    float a2 = rs * g2[c], b2 = be2[c] - mean * a2;
    float a5 = rs * g5[c], b5 = be5[c] - mean * a5;
    for (int m = 0; m < M; ++m) {
        float v = nodes[m * C + c];
        theta0[m * C + c] = v * a2 + b2;
        gmat[m * C + c]   = v * a5 + b5;
    }
}

// ---------------------------------------------------------------------------
// k3: theta1 = theta0 @ w_lin^T + b_lin   (128x1024 @ 1024x1024)
// One wave per 16x16 output tile, K-loop of 32 with bf16 WMMA.
// Operand layouts used here (not the ISA k order): A lane=row M, K map {0-15}/{16-31};
//              B lane=col N, K map {0-15}/{16-31}; D vgpr e -> M=e+8*hi, N=lane&15
// ---------------------------------------------------------------------------
__global__ void __launch_bounds__(32)
k3_linear_wmma(const float* __restrict__ theta0, const float* __restrict__ wlin,
               const float* __restrict__ blin, float* __restrict__ theta1) {
    int lane = threadIdx.x;
    int tile = blockIdx.x;        // 8 row-tiles x 64 col-tiles
    int rt = tile & 7;
    int qt = tile >> 3;
    int l15 = lane & 15, hi = lane >> 4;
    int arow = rt * 16 + l15;     // theta0 row (m)
    int bcol = qt * 16 + l15;     // w_lin row j = output column
    v8f acc = {};
    for (int k0 = 0; k0 < C; k0 += 32) {
        v16bf a, b;
        int kaBase = k0 + hi * 16;
        #pragma unroll
        for (int idx = 0; idx < 16; ++idx) {
            int ka = kaBase + idx;
            a[idx] = (__bf16)theta0[arow * C + ka];
        }
        int kb = k0 + hi * 16;
        #pragma unroll
        for (int idx = 0; idx < 16; ++idx)
            b[idx] = (__bf16)wlin[bcol * C + kb + idx];
        acc = __builtin_amdgcn_wmma_f32_16x16x32_bf16(false, a, false, b,
                                                      (short)0, acc, false, false);
    }
    float bl = blin[bcol];
    #pragma unroll
    for (int e = 0; e < 8; ++e) {
        int outrow = rt * 16 + e + hi * 8;
        theta1[outrow * C + bcol] = acc[e] + bl;
    }
}

// ---------------------------------------------------------------------------
// k4: bn3 per node-row of theta1, then fold bn1 scale:
//   thetaP[m,c] = bf16( bn3(theta1)[m,c] * a1_c ),  biasM[m] = sum_c bn3*b1_c
// ---------------------------------------------------------------------------
__global__ void k4_theta_final(const float* __restrict__ theta1,
                               const float* __restrict__ g3, const float* __restrict__ be3,
                               const float* __restrict__ a1, const float* __restrict__ b1,
                               __bf16* __restrict__ thetaP, float* __restrict__ biasM) {
    __shared__ float ssum[256], ssq[256];
    int m = blockIdx.x, tid = threadIdx.x;
    float s = 0.f, q = 0.f;
    for (int c = tid; c < C; c += 256) { float v = theta1[m * C + c]; s += v; q += v * v; }
    ssum[tid] = s; ssq[tid] = q; __syncthreads();
    for (int off = 128; off > 0; off >>= 1) {
        if (tid < off) { ssum[tid] += ssum[tid + off]; ssq[tid] += ssq[tid + off]; }
        __syncthreads();
    }
    float mean = ssum[0] / (float)C;
    float var  = ssq[0]  / (float)C - mean * mean;
    float a3 = rsqrtf(var + EPS) * g3[m];
    float b3 = be3[m] - mean * a3;
    __syncthreads();
    float bacc = 0.f;
    for (int c = tid; c < C; c += 256) {
        float t2 = theta1[m * C + c] * a3 + b3;
        thetaP[m * C + c] = (__bf16)(t2 * a1[c]);
        bacc += t2 * b1[c];
    }
    ssum[tid] = bacc; __syncthreads();
    for (int off = 128; off > 0; off >>= 1) {
        if (tid < off) ssum[tid] += ssum[tid + off];
        __syncthreads();
    }
    if (tid == 0) biasM[m] = ssum[0];
}

// ---------------------------------------------------------------------------
// k5: the big GEMM. f[s, m] = sum_c x[n,c,s'] * thetaP[m,c] + biasM[m]
// 4 waves/block; each wave owns one 16-row s-tile and all 8 m-tiles
// (8 f32 accumulators). A loaded from raw input (bn1 folded; coalesced 64B
// per 16-lane group), B is contiguous 32B v16bf loads from the L2-resident
// 256KB thetaP.  802,816 v_wmma_f32_16x16x32_bf16 total.
// ---------------------------------------------------------------------------
__global__ void __launch_bounds__(128)
k5_big_gemm(const float* __restrict__ x, const __bf16* __restrict__ thetaP,
            const float* __restrict__ biasM, float* __restrict__ f) {
    int lane = threadIdx.x & 31;
    int wave = threadIdx.x >> 5;
    int stg  = blockIdx.x * 4 + wave;      // 0..3135 s-tiles (32 n * 98 tiles)
    int n     = stg / 98;
    int stile = stg % 98;
    int s0 = stile * 16;
    int l15 = lane & 15, hi = lane >> 4;
    const float* xn = x + (size_t)n * C * SPN;
    int srow = s0 + l15;
    v8f acc[8] = {};
    for (int k0 = 0; k0 < C; k0 += 32) {
        v16bf a;
        int kaBase = k0 + hi * 16;
        #pragma unroll
        for (int idx = 0; idx < 16; ++idx) {
            int ka = kaBase + idx;   // channel index
            a[idx] = (__bf16)xn[(size_t)ka * SPN + srow];
        }
        int kb = k0 + hi * 16;
        #pragma unroll
        for (int mt = 0; mt < 8; ++mt) {
            int mrow = mt * 16 + l15;
            v16bf b = *(const v16bf*)(thetaP + mrow * C + kb);  // 32B aligned
            acc[mt] = __builtin_amdgcn_wmma_f32_16x16x32_bf16(false, a, false, b,
                                                              (short)0, acc[mt], false, false);
        }
    }
    size_t sg = (size_t)n * SPN + s0;
    #pragma unroll
    for (int mt = 0; mt < 8; ++mt) {
        int mcol = mt * 16 + l15;
        float bm = biasM[mcol];
        #pragma unroll
        for (int e = 0; e < 8; ++e) {
            int sr = e + hi * 8;
            f[(sg + sr) * M + mcol] = acc[mt][e] + bm;
        }
    }
}

// ---------------------------------------------------------------------------
// k6a/k6b: deterministic two-stage bn4 stats over all 50176 rows per node
// ---------------------------------------------------------------------------
__global__ void k6a_f_partial(const float* __restrict__ f,
                              float* __restrict__ psum, float* __restrict__ psq) {
    __shared__ float sh[256], sh2[256];
    int b = blockIdx.x, tid = threadIdx.x;
    int m = tid & 127, half = tid >> 7;
    float s = 0.f, q = 0.f;
    for (int i = half; i < 256; i += 2) {
        float v = f[((size_t)b * 256 + i) * M + m];
        s += v; q += v * v;
    }
    sh[tid] = s; sh2[tid] = q; __syncthreads();
    if (tid < 128) {
        psum[b * M + m] = sh[tid] + sh[tid + 128];
        psq [b * M + m] = sh2[tid] + sh2[tid + 128];
    }
}

__global__ void k6b_f_final(const float* __restrict__ psum, const float* __restrict__ psq,
                            const float* __restrict__ g4, const float* __restrict__ be4,
                            float* __restrict__ A4, float* __restrict__ B4) {
    int m = threadIdx.x;   // 128 threads
    float s = 0.f, q = 0.f;
    for (int b = 0; b < STOT / 256; ++b) { s += psum[b * M + m]; q += psq[b * M + m]; }
    float mean = s / (float)STOT;
    float var  = q / (float)STOT - mean * mean;
    float a = rsqrtf(var + EPS) * g4[m];
    A4[m] = a;
    B4[m] = be4[m] - mean * a;
}

// ---------------------------------------------------------------------------
// k7: softmax->argmax == argmax of bn4(f). Histogram the winners (int atomics
// are deterministic); bn6 stats then follow from the histogram alone.
// ---------------------------------------------------------------------------
__global__ void k7zero(int* __restrict__ counts) { counts[threadIdx.x] = 0; }

__global__ void k7a_argmax(const float* __restrict__ f, const float* __restrict__ A4,
                           const float* __restrict__ B4, int* __restrict__ idx,
                           int* __restrict__ counts) {
    __shared__ float sA[128], sB[128];
    __shared__ int hist[128];
    int tid = threadIdx.x;
    if (tid < 128) { sA[tid] = A4[tid]; sB[tid] = B4[tid]; hist[tid] = 0; }
    __syncthreads();
    int s = blockIdx.x * 256 + tid;
    const float4* row = (const float4*)(f + (size_t)s * M);   // 512B aligned row
    float best = -3.4e38f; int bi = 0;
    #pragma unroll 4
    for (int m4 = 0; m4 < M / 4; ++m4) {
        float4 v4 = row[m4];
        int m = m4 * 4;
        float v;
        v = v4.x * sA[m + 0] + sB[m + 0]; if (v > best) { best = v; bi = m + 0; }
        v = v4.y * sA[m + 1] + sB[m + 1]; if (v > best) { best = v; bi = m + 1; }
        v = v4.z * sA[m + 2] + sB[m + 2]; if (v > best) { best = v; bi = m + 2; }
        v = v4.w * sA[m + 3] + sB[m + 3]; if (v > best) { best = v; bi = m + 3; }
    }
    idx[s] = bi;
    atomicAdd(&hist[bi], 1);
    __syncthreads();
    if (tid < 128) atomicAdd(&counts[tid], hist[tid]);
}

__global__ void k7b_ztable(const float* __restrict__ gmat, const int* __restrict__ counts,
                           const float* __restrict__ g6, const float* __restrict__ be6,
                           float* __restrict__ z) {
    int c = blockIdx.x * blockDim.x + threadIdx.x;
    if (c >= C) return;
    float inv = 1.f / (float)STOT;
    float s = 0.f, q = 0.f;
    for (int m = 0; m < M; ++m) {
        float p = (float)counts[m] * inv;
        float g = gmat[m * C + c];
        s += p * g; q += p * g * g;
    }
    float var = q - s * s;
    float a6 = rsqrtf(var + EPS) * g6[c];
    float b6 = be6[c] - s * a6;
    for (int m = 0; m < M; ++m) {
        float y = gmat[m * C + c] * a6 + b6;
        z[m * C + c] = (y >= 0.f) ? y : NEG_SLOPE * y;   // leaky fused into table
    }
}

// ---------------------------------------------------------------------------
// k8: out[n,c,s'] = z[idx[n,s'], c] — gather writeback of the 205MB output.
// SPN % 4 == 0, so 4 consecutive s' share (n,c): one int4 idx load, four
// 4B table gathers (512KB z stays in L2), one b128 store per thread.
// ---------------------------------------------------------------------------
__global__ void k8_out(const float* __restrict__ z, const int* __restrict__ idx,
                       float* __restrict__ out) {
    unsigned g4id = blockIdx.x * 256u + threadIdx.x;    // < 12,845,056
    unsigned sp4 = g4id % (unsigned)(SPN / 4);
    unsigned nc  = g4id / (unsigned)(SPN / 4);
    unsigned c   = nc % (unsigned)C;
    unsigned n   = nc / (unsigned)C;
    int4 i4 = *(const int4*)(idx + n * SPN + sp4 * 4);  // 16B aligned
    float4 o;
    o.x = z[(size_t)i4.x * C + c];
    o.y = z[(size_t)i4.y * C + c];
    o.z = z[(size_t)i4.z * C + c];
    o.w = z[(size_t)i4.w * C + c];
    *(float4*)(out + (size_t)g4id * 4) = o;
}

// ---------------------------------------------------------------------------
extern "C" void kernel_launch(void* const* d_in, const int* in_sizes, int n_in,
                              void* d_out, int out_size, void* d_ws, size_t ws_size,
                              hipStream_t stream) {
    (void)in_sizes; (void)n_in; (void)out_size; (void)ws_size;
    const float* x     = (const float*)d_in[0];
    const float* nodes = (const float*)d_in[1];
    const float* wlin  = (const float*)d_in[2];
    const float* blin  = (const float*)d_in[3];
    const float* g1  = (const float*)d_in[4];
    const float* be1 = (const float*)d_in[5];
    const float* g2  = (const float*)d_in[6];
    const float* be2 = (const float*)d_in[7];
    const float* g3  = (const float*)d_in[8];
    const float* be3 = (const float*)d_in[9];
    const float* g4  = (const float*)d_in[10];
    const float* be4 = (const float*)d_in[11];
    const float* g5  = (const float*)d_in[12];
    const float* be5 = (const float*)d_in[13];
    const float* g6  = (const float*)d_in[14];
    const float* be6 = (const float*)d_in[15];
    float* out = (float*)d_out;

    char* ws = (char*)d_ws;
    auto carve = [&ws](size_t bytes) -> char* {
        char* p = ws;
        ws += (bytes + 255) & ~(size_t)255;
        return p;
    };
    float*  a1     = (float*) carve(C * 4);
    float*  b1     = (float*) carve(C * 4);
    float*  theta0 = (float*) carve((size_t)M * C * 4);
    float*  gmat   = (float*) carve((size_t)M * C * 4);
    float*  theta1 = (float*) carve((size_t)M * C * 4);
    __bf16* thetaP = (__bf16*)carve((size_t)M * C * 2);
    float*  biasM  = (float*) carve(M * 4);
    float*  psum   = (float*) carve((size_t)(STOT / 256) * M * 4);
    float*  psq    = (float*) carve((size_t)(STOT / 256) * M * 4);
    float*  A4     = (float*) carve(M * 4);
    float*  B4     = (float*) carve(M * 4);
    int*    counts = (int*)   carve(M * 4);
    float*  ztab   = (float*) carve((size_t)M * C * 4);
    int*    idx    = (int*)   carve((size_t)STOT * 4);
    float*  fbuf   = (float*) carve((size_t)STOT * M * 4);   // 25.7 MB

    k1_input_stats<<<C, 256, 0, stream>>>(x, g1, be1, a1, b1);
    k2_nodes<<<C / 256, 256, 0, stream>>>(nodes, g2, be2, g5, be5, theta0, gmat);
    k3_linear_wmma<<<(M / 16) * (C / 16), 32, 0, stream>>>(theta0, wlin, blin, theta1);
    k4_theta_final<<<M, 256, 0, stream>>>(theta1, g3, be3, a1, b1, thetaP, biasM);
    k5_big_gemm<<<(Nn * (SPN / 16)) / 4, 128, 0, stream>>>(x, thetaP, biasM, fbuf);
    k6a_f_partial<<<STOT / 256, 256, 0, stream>>>(fbuf, psum, psq);
    k6b_f_final<<<1, 128, 0, stream>>>(psum, psq, g4, be4, A4, B4);
    k7zero<<<1, 128, 0, stream>>>(counts);
    k7a_argmax<<<STOT / 256, 256, 0, stream>>>(fbuf, A4, B4, idx, counts);
    k7b_ztable<<<C / 256, 256, 0, stream>>>(gmat, counts, g6, be6, ztab);
    k8_out<<<(unsigned)((size_t)Nn * C * SPN / (4 * 256)), 256, 0, stream>>>(ztab, idx, out);
}
